// WSDDN_avg_57947698757637
// MI455X (gfx1250) — matrix operand order repaired and checked by reading, not compiled, not run on hardware
//
#include <hip/hip_runtime.h>
#include <math.h>

// ---------------- problem constants ----------------
#define BB 8
#define CC 512
#define TT 8192
#define PP 512
#define KK 30
#define ROWS (BB * PP)   // 4096

typedef unsigned short ushort_t;
typedef __attribute__((ext_vector_type(16))) __bf16 v16bf;
typedef __attribute__((ext_vector_type(8)))  __bf16 bf16x8;
typedef __attribute__((ext_vector_type(8)))  float  v8f;

union FragAB { v16bf v; bf16x8 h[2]; };

__device__ __forceinline__ ushort_t f32_to_bf16(float f) {
  unsigned u = __float_as_uint(f);
  unsigned r = u + 0x7FFFu + ((u >> 16) & 1u);   // round-to-nearest-even
  return (ushort_t)(r >> 16);
}

// ---------------- weight prep: f32 -> bf16, fuse heads into padded N=64 ----------------
__global__ void prep_kernel(const float* __restrict__ W6, const float* __restrict__ W7,
                            const float* __restrict__ Wc, const float* __restrict__ bc,
                            const float* __restrict__ Wd, const float* __restrict__ bd,
                            ushort_t* __restrict__ W6b, ushort_t* __restrict__ W7b,
                            ushort_t* __restrict__ Wcdb, float* __restrict__ bias64) {
  const int n6 = 1024 * 512, n7 = 512 * 1024, nh = 64 * 512;
  int stride = gridDim.x * blockDim.x;
  for (int idx = blockIdx.x * blockDim.x + threadIdx.x; idx < n6 + n7 + nh; idx += stride) {
    if (idx < n6) {
      W6b[idx] = f32_to_bf16(W6[idx]);
    } else if (idx < n6 + n7) {
      int t = idx - n6;
      W7b[t] = f32_to_bf16(W7[t]);
    } else {
      int t = idx - n6 - n7;
      int row = t >> 9, col = t & 511;
      float v = 0.0f;
      if (row < KK) v = Wc[row * 512 + col];
      else if (row >= 32 && row < 32 + KK) v = Wd[(row - 32) * 512 + col];
      Wcdb[t] = f32_to_bf16(v);
    }
  }
  int i = blockIdx.x * blockDim.x + threadIdx.x;
  if (i < 64) {
    float v = 0.0f;
    if (i < KK) v = bc[i];
    else if (i >= 32 && i < 32 + KK) v = bd[i - 32];
    bias64[i] = v;
  }
}

// ---------------- segment-mean pooling (x is L2-resident: 134MB < 192MB) ----------------
// one block per (b,p); one wave per channel; lanes strided over t (coalesced)
__global__ void pool_kernel(const float* __restrict__ x, const int* __restrict__ boxes,
                            ushort_t* __restrict__ featsB) {
  int bp = blockIdx.x;            // 0..4095
  int b  = bp >> 9;
  int s  = boxes[bp * 2 + 0];
  int e  = boxes[bp * 2 + 1];
  float inv_len = 1.0f / (float)(e - s);
  int lane = threadIdx.x & 31;
  int wave = threadIdx.x >> 5;    // 0..7
  for (int c = wave; c < CC; c += 8) {
    const float* row = x + ((size_t)b * CC + c) * TT;
    float sum = 0.0f;
    for (int t = s + lane; t < e; t += 32) sum += row[t];
#pragma unroll
    for (int o = 16; o > 0; o >>= 1) sum += __shfl_down(sum, o);
    if (lane == 0) featsB[(size_t)bp * CC + c] = f32_to_bf16(sum * inv_len);
  }
}

// ---------------- bf16 WMMA GEMM: out[m,n] = act( sum_k A[m,k]*Bw[n,k] + bias[n] ) ----
// Fragment layouts per CDNA5 ISA 7.12.2 (wave32):
//  A 16x32: lane m=l&15; chunks K = half*8..+7 and half*8+16..+23  (half = l>>4)
//  B 32x16: lane n=l&15; chunks K = half*8..+7 and half*8+16..+23  (same as A)
//  D 16x16: vgpr r -> M = r + half*8, N = l&15
__global__ void gemm_wmma_bf16(const ushort_t* __restrict__ A, const ushort_t* __restrict__ Bw,
                               const float* __restrict__ bias,
                               float* __restrict__ outF, ushort_t* __restrict__ outB,
                               float* __restrict__ clsOut, float* __restrict__ detOut,
                               int M, int N, int K, int relu) {
  int wid   = (blockIdx.x * blockDim.x + threadIdx.x) >> 5;
  int tilesN = N >> 5;
  int tm = wid / tilesN, tn = wid - tm * tilesN;
  int m0 = tm << 5, n0 = tn << 5;
  int lane  = threadIdx.x & 31;
  int idx16 = lane & 15;
  int half  = lane >> 4;

  const ushort_t* a0 = A  + (size_t)(m0 + idx16) * K + half * 8;
  const ushort_t* a1 = a0 + (size_t)16 * K;
  const ushort_t* b0 = Bw + (size_t)(n0 + idx16) * K + half * 8;
  const ushort_t* b1 = b0 + (size_t)16 * K;

  v8f acc[2][2] = {};
  for (int k = 0; k < K; k += 32) {
    __builtin_prefetch(a0 + k + 256, 0, 3);   // global_prefetch_b8, stream next tiles
    __builtin_prefetch(b0 + k + 256, 0, 3);
    FragAB fa[2], fb[2];
    fa[0].h[0] = *(const bf16x8*)(const void*)(a0 + k);
    fa[0].h[1] = *(const bf16x8*)(const void*)(a0 + k + 16);
    fa[1].h[0] = *(const bf16x8*)(const void*)(a1 + k);
    fa[1].h[1] = *(const bf16x8*)(const void*)(a1 + k + 16);
    fb[0].h[0] = *(const bf16x8*)(const void*)(b0 + k);
    fb[0].h[1] = *(const bf16x8*)(const void*)(b0 + k + 16);
    fb[1].h[0] = *(const bf16x8*)(const void*)(b1 + k);
    fb[1].h[1] = *(const bf16x8*)(const void*)(b1 + k + 16);
#pragma unroll
    for (int i = 0; i < 2; ++i)
#pragma unroll
      for (int j = 0; j < 2; ++j)
        acc[i][j] = __builtin_amdgcn_wmma_f32_16x16x32_bf16(
            false, fa[i].v, false, fb[j].v, (short)0, acc[i][j], false, false);
  }

#pragma unroll
  for (int i = 0; i < 2; ++i) {
#pragma unroll
    for (int j = 0; j < 2; ++j) {
#pragma unroll
      for (int r = 0; r < 8; ++r) {
        int m = m0 + i * 16 + half * 8 + r;
        int n = n0 + j * 16 + idx16;
        float v = acc[i][j][r] + bias[n];
        if (relu) v = fmaxf(v, 0.0f);
        if (outF) outF[(size_t)m * N + n] = v;
        if (outB) outB[(size_t)m * N + n] = f32_to_bf16(v);
        if (clsOut) {
          if (n < KK)                        clsOut[(size_t)m * KK + n]        = v;
          else if (n >= 32 && n < 32 + KK)   detOut[(size_t)m * KK + (n - 32)] = v;
        }
      }
    }
  }
}

// ---------------- softmax stats over K=30 (one wave per row) ----------------
__global__ void class_stats_kernel(const float* __restrict__ cls,
                                   float* __restrict__ rmax, float* __restrict__ rsum) {
  int row  = (blockIdx.x * blockDim.x + threadIdx.x) >> 5;
  int lane = threadIdx.x & 31;
  float v = (lane < KK) ? cls[(size_t)row * KK + lane] : -INFINITY;
  float m = v;
#pragma unroll
  for (int o = 16; o > 0; o >>= 1) m = fmaxf(m, __shfl_down(m, o));
  m = __shfl(m, 0);
  float e = (lane < KK) ? expf(v - m) : 0.0f;
#pragma unroll
  for (int o = 16; o > 0; o >>= 1) e += __shfl_down(e, o);
  if (lane == 0) { rmax[row] = m; rsum[row] = e; }
}

// ---------------- softmax stats over P=512 (one block per (b,k)) ----------------
__global__ void det_stats_kernel(const float* __restrict__ det,
                                 float* __restrict__ dmax, float* __restrict__ dsum) {
  int bk = blockIdx.x;                 // 0..239
  int b = bk / KK, k = bk - b * KK;
  int tid = threadIdx.x;               // 256
  const float* base = det + (size_t)b * PP * KK + k;
  float v0 = base[(size_t)tid * KK];
  float v1 = base[(size_t)(tid + 256) * KK];
  __shared__ float sm[256];
  sm[tid] = fmaxf(v0, v1);
  __syncthreads();
  for (int o = 128; o > 0; o >>= 1) { if (tid < o) sm[tid] = fmaxf(sm[tid], sm[tid + o]); __syncthreads(); }
  float m = sm[0];
  __syncthreads();
  sm[tid] = expf(v0 - m) + expf(v1 - m);
  __syncthreads();
  for (int o = 128; o > 0; o >>= 1) { if (tid < o) sm[tid] += sm[tid + o]; __syncthreads(); }
  if (tid == 0) { dmax[bk] = m; dsum[bk] = sm[0]; }
}

// ---------------- joint prob + video prob (one block per (b,k)) ----------------
__global__ void finalize_kernel(const float* __restrict__ cls, const float* __restrict__ det,
                                const float* __restrict__ rmax, const float* __restrict__ rsum,
                                const float* __restrict__ dmax, const float* __restrict__ dsum,
                                float* __restrict__ joint, float* __restrict__ video) {
  int bk = blockIdx.x;
  int b = bk / KK, k = bk - b * KK;
  int tid = threadIdx.x;
  float dm = dmax[bk];
  float dsInv = 1.0f / dsum[bk];
  __shared__ float sm[256];
  float local = 0.0f;
  for (int p = tid; p < PP; p += 256) {
    size_t row = (size_t)b * PP + p;
    float cp = expf(cls[row * KK + k] - rmax[row]) / rsum[row];
    float dp = expf(det[row * KK + k] - dm) * dsInv;
    float jv = cp * dp;
    joint[row * KK + k] = jv;
    local += jv;
  }
  sm[tid] = local;
  __syncthreads();
  for (int o = 128; o > 0; o >>= 1) { if (tid < o) sm[tid] += sm[tid + o]; __syncthreads(); }
  if (tid == 0) video[bk] = sm[0];
}

// ---------------- launch ----------------
extern "C" void kernel_launch(void* const* d_in, const int* in_sizes, int n_in,
                              void* d_out, int out_size, void* d_ws, size_t ws_size,
                              hipStream_t stream) {
  (void)in_sizes; (void)n_in; (void)out_size; (void)ws_size;
  const float* x    = (const float*)d_in[0];
  const int*  boxes = (const int*) d_in[1];
  const float* W6   = (const float*)d_in[2];
  const float* b6   = (const float*)d_in[3];
  const float* W7   = (const float*)d_in[4];
  const float* b7   = (const float*)d_in[5];
  const float* Wc   = (const float*)d_in[6];
  const float* bc   = (const float*)d_in[7];
  const float* Wd   = (const float*)d_in[8];
  const float* bd   = (const float*)d_in[9];

  float* out = (float*)d_out;
  const size_t PK = (size_t)ROWS * KK;             // 122880
  float* o_video = out;                            // [B,K]      240
  float* o_joint = out + 240;                      // [B,P,K]
  float* o_class = out + 240 + PK;                 // [B,P,K]
  float* o_det   = out + 240 + 2 * PK;             // [B,P,K]
  float* o_h7    = out + 240 + 3 * PK;             // [B,P,512]

  char* ws = (char*)d_ws;
  size_t off = 0;
  auto alloc = [&](size_t bytes) -> void* {
    void* p = ws + off;
    off = (off + bytes + 255) & ~(size_t)255;
    return p;
  };
  ushort_t* featsB = (ushort_t*)alloc((size_t)ROWS * 512 * 2);   // 4 MB
  ushort_t* W6b    = (ushort_t*)alloc((size_t)1024 * 512 * 2);   // 1 MB
  ushort_t* W7b    = (ushort_t*)alloc((size_t)512 * 1024 * 2);   // 1 MB
  ushort_t* Wcdb   = (ushort_t*)alloc((size_t)64 * 512 * 2);     // 64 KB
  float*    bias64 = (float*)   alloc(64 * 4);
  ushort_t* h6b    = (ushort_t*)alloc((size_t)ROWS * 1024 * 2);  // 8 MB
  ushort_t* h7b    = (ushort_t*)alloc((size_t)ROWS * 512 * 2);   // 4 MB
  float*    rmax   = (float*)   alloc((size_t)ROWS * 4);
  float*    rsum   = (float*)   alloc((size_t)ROWS * 4);
  float*    dmax   = (float*)   alloc(240 * 4);
  float*    dsum   = (float*)   alloc(240 * 4);

  // 1. weights -> bf16 (+ fused padded heads)
  prep_kernel<<<512, 256, 0, stream>>>(W6, W7, Wc, bc, Wd, bd, W6b, W7b, Wcdb, bias64);
  // 2. segment-mean pooling -> bf16 feats [4096,512]
  pool_kernel<<<ROWS, 256, 0, stream>>>(x, boxes, featsB);
  // 3. h6 = relu(feats @ W6^T + b6)  [4096,1024], bf16 out only
  gemm_wmma_bf16<<<512, 256, 0, stream>>>(featsB, W6b, b6, nullptr, h6b,
                                          nullptr, nullptr, ROWS, 1024, 512, 1);
  // 4. h7 = relu(h6 @ W7^T + b7)  [4096,512], f32 -> d_out, bf16 -> ws
  gemm_wmma_bf16<<<256, 256, 0, stream>>>(h6b, W7b, b7, o_h7, h7b,
                                          nullptr, nullptr, ROWS, 512, 1024, 1);
  // 5. fused heads: [4096,64] -> class_logits / det_logits (N padded 30->32 each)
  gemm_wmma_bf16<<<32, 256, 0, stream>>>(h7b, Wcdb, bias64, nullptr, nullptr,
                                         o_class, o_det, ROWS, 64, 512, 0);
  // 6. softmax stats
  class_stats_kernel<<<ROWS / 8, 256, 0, stream>>>(o_class, rmax, rsum);
  det_stats_kernel<<<BB * KK, 256, 0, stream>>>(o_det, dmax, dsum);
  // 7. joint + video
  finalize_kernel<<<BB * KK, 256, 0, stream>>>(o_class, o_det, rmax, rsum,
                                               dmax, dsum, o_joint, o_video);
}
